// QuadConv_16458314678313
// MI455X (gfx1250) — matrix operand order repaired and checked by reading, not compiled, not run on hardware
//
#include <hip/hip_runtime.h>
#include <hip/hip_bf16.h>

typedef __attribute__((ext_vector_type(16))) __bf16 v16bf;
typedef __attribute__((ext_vector_type(8)))  __bf16 v8bf;
typedef __attribute__((ext_vector_type(8)))  float  v8f;
typedef __attribute__((ext_vector_type(4)))  float  v4f;

#define C_IN   128
#define C_OUT  128
#define KNBR   9
#define FAN_IN (KNBR * C_IN)   // 1152

// ---------------------------------------------------------------------------
// Prep: W [C_OUT][FAN_IN] fp32 -> bf16, same row-major layout (288 KB in ws).
// W row-major [n][k] is B^T: a lane's WMMA B-fragment (B[kb..kb+7,kb+16..kb+23][n] =
// W[n][same k]) is two contiguous 16-byte runs.
// ---------------------------------------------------------------------------
__global__ void wcvt_kernel(const float* __restrict__ W,
                            __bf16* __restrict__ Wb, int n) {
    int i = blockIdx.x * blockDim.x + threadIdx.x;
    if (i < n) Wb[i] = (__bf16)W[i];
}

struct AFrag { v16bf v[2]; };          // 2 M-tiles
struct BFrag { v16bf v[4]; };          // 4 N-tiles
struct RowCtx {                        // per-neighbor-slot gather context
    const float* ap[2];                // row base (+half*8 folded in)
    float        mask[2];              // 0.0 for -1 neighbors
};

// Once per neighbor slot j: load the two row indices, build base pointers.
__device__ __forceinline__ RowCtx
load_rows(const float* __restrict__ feat,
          const long long* __restrict__ i0,
          const long long* __restrict__ i1,
          int j, int half) {
    RowCtx rc;
    const long long r0 = i0[j];
    const long long r1 = i1[j];
    rc.mask[0] = (r0 >= 0) ? 1.0f : 0.0f;
    rc.mask[1] = (r1 >= 0) ? 1.0f : 0.0f;
    rc.ap[0] = feat + (r0 >= 0 ? r0 : 0) * (long long)C_IN + half * 8;
    rc.ap[1] = feat + (r1 >= 0 ? r1 : 0) * (long long)C_IN + half * 8;
    return rc;
}

// A fragment for one 32-wide K chunk: 4 x global_load_b128 per M-tile with
// immediate offsets, then fp32 -> bf16 (+mask) in-register.
// Lane layout (16-bit A 16x32): lanes 0-15 K {0..7,16..23}, lanes 16-31
// K {8..15,24..31} of the chunk (half*8 already folded into rc.ap).
__device__ __forceinline__ AFrag
load_a_chunk(const RowCtx& rc, int cc) {
    AFrag A;
    #pragma unroll
    for (int mt = 0; mt < 2; ++mt) {
        const float* ap = rc.ap[mt] + cc * 32;
        v4f a0 = *(const v4f*)(ap +  0);
        v4f a1 = *(const v4f*)(ap +  4);
        v4f a2 = *(const v4f*)(ap + 16);
        v4f a3 = *(const v4f*)(ap + 20);
        const float m = rc.mask[mt];
        v16bf f;
        #pragma unroll
        for (int e = 0; e < 4; ++e) {
            f[e +  0] = (__bf16)(a0[e] * m);
            f[e +  4] = (__bf16)(a1[e] * m);
            f[e +  8] = (__bf16)(a2[e] * m);
            f[e + 12] = (__bf16)(a3[e] * m);
        }
        A.v[mt] = f;
    }
    return A;
}

// B fragments for 4 N-tiles: contiguous 2x16B per lane from bf16 W.
__device__ __forceinline__ BFrag
load_b(const __bf16* __restrict__ Wb, int ks, int nbase, int l16, int half) {
    BFrag B;
    const int kb = ks * 32 + half * 8;
    #pragma unroll
    for (int t = 0; t < 4; ++t) {
        const __bf16* bp = Wb + (nbase + t * 16 + l16) * FAN_IN + kb;
        v8bf b0 = *(const v8bf*)(bp);
        v8bf b1 = *(const v8bf*)(bp + 16);
        v16bf f;
        #pragma unroll
        for (int e = 0; e < 8; ++e) { f[e] = b0[e]; f[e + 8] = b1[e]; }
        B.v[t] = f;
    }
    return B;
}

__device__ __forceinline__ void
mma_step(const AFrag& A, const BFrag& B, v8f (&acc)[2][4]) {
    #pragma unroll
    for (int mt = 0; mt < 2; ++mt)
        #pragma unroll
        for (int t = 0; t < 4; ++t)
            acc[mt][t] = __builtin_amdgcn_wmma_f32_16x16x32_bf16(
                false, A.v[mt], false, B.v[t],
                (short)0, acc[mt][t], false, false);
}

// ---------------------------------------------------------------------------
// Block = 256 threads = 8 waves tiling 128(M) x 128(N). Waves arranged
// 4(M) x 2(N): each wave owns 32 rows (2 M-tiles) x 64 channels (4 N-tiles).
// Outer loop: 9 neighbor slots; inner: 4 unrolled 32-wide K chunks, 8 WMMAs
// each. Row context and the first chunk of the next slot are pipelined one
// iteration ahead so the idx->address->gather chain never stalls the WMMAs.
// ---------------------------------------------------------------------------
__global__ void __launch_bounds__(256)
quadconv_bf16_wmma_kernel(const float* __restrict__ feat,
                          const long long* __restrict__ nidx,
                          const __bf16* __restrict__ Wb,
                          const float* __restrict__ bias,
                          float* __restrict__ out) {
    const int lane  = threadIdx.x & 31;
    const int wave  = threadIdx.x >> 5;
    const int half  = lane >> 4;
    const int l16   = lane & 15;
    const int mgrp  = wave >> 1;                 // 0..3
    const int ngrp  = wave & 1;                  // 0..1
    const int mwave = blockIdx.x * 128 + mgrp * 32;
    const int nbase = ngrp * 64;

    const long long* __restrict__ i0 = nidx + (long long)(mwave + l16) * KNBR;
    const long long* __restrict__ i1 = i0 + 16 * KNBR;

    v8f acc[2][4] = {};

    RowCtx rc = load_rows(feat, i0, i1, 0, half);
    AFrag  a0 = load_a_chunk(rc, 0);
    BFrag  b0 = load_b(Wb, 0, nbase, l16, half);

    for (int j = 0; j < KNBR; ++j) {
        const int jn = (j + 1 < KNBR) ? j + 1 : KNBR - 1;   // tail-safe
        RowCtx rn = load_rows(feat, i0, i1, jn, half);      // next slot's rows

        #pragma unroll
        for (int cc = 0; cc < 4; ++cc) {
            AFrag a = (cc == 0) ? a0 : load_a_chunk(rc, cc);
            BFrag b = (cc == 0) ? b0 : load_b(Wb, j * 4 + cc, nbase, l16, half);
            mma_step(a, b, acc);
        }

        // preload first chunk of next slot (dead + DCE'd on last iteration)
        a0 = load_a_chunk(rn, 0);
        b0 = load_b(Wb, jn * 4, nbase, l16, half);
        rc = rn;
    }

    // ---- epilogue: bias + store (coalesced per 16-lane half) ----
    // C/D layout: VGPR i -> M = i + 8*half within tile; N = l16 within tile.
    #pragma unroll
    for (int mt = 0; mt < 2; ++mt) {
        #pragma unroll
        for (int t = 0; t < 4; ++t) {
            const int n = nbase + t * 16 + l16;
            const float bv = bias[n];
            #pragma unroll
            for (int i = 0; i < 8; ++i) {
                const long long row = mwave + mt * 16 + half * 8 + i;
                out[row * (long long)C_OUT + n] = acc[mt][t][i] + bv;
            }
        }
    }
}

extern "C" void kernel_launch(void* const* d_in, const int* in_sizes, int n_in,
                              void* d_out, int out_size, void* d_ws, size_t ws_size,
                              hipStream_t stream) {
    const float*     feat = (const float*)d_in[0];      // [N, 128] f32
    const long long* nidx = (const long long*)d_in[1];  // [N, 9]   i64
    const float*     W    = (const float*)d_in[2];      // [128, 1152] f32
    const float*     bias = (const float*)d_in[3];      // [128] f32
    float*           out  = (float*)d_out;              // [N, 128] f32
    __bf16*          Wb   = (__bf16*)d_ws;              // 288 KB scratch

    const int nW = C_OUT * FAN_IN;
    wcvt_kernel<<<(nW + 255) / 256, 256, 0, stream>>>(W, Wb, nW);

    const int npts = in_sizes[0] / C_IN;                // 262144
    quadconv_bf16_wmma_kernel<<<npts / 128, 256, 0, stream>>>(
        feat, nidx, Wb, bias, out);
}
